// STGraphSAGE_31104153157825
// MI455X (gfx1250) — compile-verified, matrix operand order checked by reading, not run on hardware
//
#include <hip/hip_runtime.h>
#include <hip/hip_bf16.h>

typedef __attribute__((ext_vector_type(16))) __bf16 v16bf;
typedef __attribute__((ext_vector_type(8)))  float  v8f;

#define N_NODES  50000
#define T_STEPS  8
#define C_IN     32
#define HDIM     128
#define C_OUT    32
#define N_EDGES  800000

// B-fragment table (each frag = 32 lanes x 8 dwords = 16 bf16/lane, K=32 chunk)
#define FRAG_WL    0    // 8 frags  : Wl^T  [32 x 128]
#define FRAG_WR    8    // 8 frags  : Wr^T  [32 x 128]
#define FRAG_WIH   16   // 96 frags : W_ih^T [128 x 384] (24 col-tiles x 4 K-chunks)
#define FRAG_WHH   112  // 96 frags : W_hh^T
#define FRAG_WREC  208  // 8 frags  : W_rec^T [128 x 32]
#define FRAG_WC1   216  // 16 frags : W_c1^T  [128 x 64]
#define N_FRAGS    232

__device__ __forceinline__ v8f wmma_bf16(v16bf a, v16bf b, v8f c) {
  return __builtin_amdgcn_wmma_f32_16x16x32_bf16(false, a, false, b, (short)0, c,
                                                 false, false);
}

__device__ __forceinline__ v16bf load_frag(const int* __restrict__ packed, int frag, int lane) {
  union { int4 i4[2]; v16bf v; } u;
  const int4* p = reinterpret_cast<const int4*>(packed + (((size_t)frag << 5) + lane) * 8);
  u.i4[0] = p[0];
  u.i4[1] = p[1];
  return u.v;
}

// A-fragment (16x32 bf16) from a row-major f32 row pointer.
// ISA layout: lane m=l&15, half=l>>4; element e -> k = (e&7) + 16*(e>>3) + 8*half
__device__ __forceinline__ v16bf load_a_g(const float* __restrict__ row, int hsel, float scale) {
  v16bf a;
  const float* p0 = row + 8 * hsel;
  const float* p1 = p0 + 16;
#pragma unroll
  for (int i = 0; i < 8; ++i) a[i] = (__bf16)(p0[i] * scale);
#pragma unroll
  for (int i = 0; i < 8; ++i) a[i + 8] = (__bf16)(p1[i] * scale);
  return a;
}

// A-fragment K-chunk kc (of 128) from a bf16 LDS row
__device__ __forceinline__ v16bf load_a_lds(const __bf16* row, int hsel, int kc) {
  v16bf a;
  const int b0 = kc * 32 + 8 * hsel;
#pragma unroll
  for (int i = 0; i < 8; ++i) a[i] = row[b0 + i];
#pragma unroll
  for (int i = 0; i < 8; ++i) a[i + 8] = row[b0 + 16 + i];
  return a;
}

__device__ __forceinline__ float sigmoidf_(float x) {
  return 1.0f / (1.0f + __expf(-x));
}

// ---------------- aux kernels ----------------

__global__ void zero_kernel(float* __restrict__ p, long long count) {
  long long i = (long long)blockIdx.x * blockDim.x + threadIdx.x;
  long long stride = (long long)gridDim.x * blockDim.x;
  for (; i < count; i += stride) p[i] = 0.0f;
}

__global__ void deg_kernel(const long long* __restrict__ ei, float* __restrict__ deg, int E) {
  int e = blockIdx.x * blockDim.x + threadIdx.x;
  if (e >= E) return;
  int d = (int)ei[(size_t)E + e];
  atomicAdd(&deg[d], 1.0f);
}

// thread handles one edge x 8 channels, loops over all T
__global__ void agg_kernel(const float* __restrict__ x, const long long* __restrict__ ei,
                           float* __restrict__ agg, int E) {
  int tid = blockIdx.x * blockDim.x + threadIdx.x;
  int e = tid >> 2;
  int g = tid & 3;
  if (e >= E) return;
  long long s = ei[e];
  long long d = ei[(size_t)E + e];
  const float* xs = x + (size_t)s * (T_STEPS * C_IN) + g * 8;
  float* ag = agg + (size_t)d * C_IN + g * 8;
#pragma unroll
  for (int t = 0; t < T_STEPS; ++t) {
    const float4* xp = reinterpret_cast<const float4*>(xs + t * C_IN);
    float4 v0 = xp[0];
    float4 v1 = xp[1];
    float* ap = ag + (size_t)t * N_NODES * C_IN;
    atomicAdd(ap + 0, v0.x); atomicAdd(ap + 1, v0.y);
    atomicAdd(ap + 2, v0.z); atomicAdd(ap + 3, v0.w);
    atomicAdd(ap + 4, v1.x); atomicAdd(ap + 5, v1.y);
    atomicAdd(ap + 6, v1.z); atomicAdd(ap + 7, v1.w);
  }
}

// convert f32 weights -> packed bf16 B-fragments in ISA VGPR layout
__global__ void prepack_kernel(const float* __restrict__ Wl, const float* __restrict__ Wr,
                               const float* __restrict__ Wih, const float* __restrict__ Whh,
                               const float* __restrict__ Wrec, const float* __restrict__ Wc1,
                               int* __restrict__ packed) {
  int idx = blockIdx.x * blockDim.x + threadIdx.x;
  if (idx >= N_FRAGS * 32) return;
  int frag = idx >> 5;
  int lane = idx & 31;
  int nn = lane & 15;
  int hsel = lane >> 4;
  const float* W;
  int tile, kc, ld;
  if (frag < FRAG_WR)         { W = Wl;   tile = frag;              kc = 0;               ld = C_IN; }
  else if (frag < FRAG_WIH)   { W = Wr;   tile = frag - FRAG_WR;    kc = 0;               ld = C_IN; }
  else if (frag < FRAG_WHH)   { int f = frag - FRAG_WIH;  W = Wih;  tile = f >> 2; kc = f & 3; ld = HDIM; }
  else if (frag < FRAG_WREC)  { int f = frag - FRAG_WHH;  W = Whh;  tile = f >> 2; kc = f & 3; ld = HDIM; }
  else if (frag < FRAG_WC1)   { int f = frag - FRAG_WREC; W = Wrec; tile = f >> 2; kc = f & 3; ld = HDIM; }
  else                        { int f = frag - FRAG_WC1;  W = Wc1;  tile = f >> 2; kc = f & 3; ld = HDIM; }
  int ng = tile * 16 + nn; // global output column (row of W)
  union { int out[8]; __bf16 v[16]; } u;
#pragma unroll
  for (int e = 0; e < 16; ++e) {
    int k = kc * 32 + (e & 7) + 16 * (e >> 3) + 8 * hsel;
    u.v[e] = (__bf16)W[(size_t)ng * ld + k]; // B[k][n] = W^T
  }
  int* dst = packed + (size_t)idx * 8;
#pragma unroll
  for (int i = 0; i < 8; ++i) dst[i] = u.out[i];
}

// ---------------- fused ST-GraphSAGE-GRU kernel ----------------
// 1 wave per 16-node tile, all T=8 steps iterated locally (h in per-wave LDS).
// Output-tile loops are deliberately NOT unrolled to keep VGPR pressure low
// (live set: a_sp/a_h fragments = 64 VGPRs + 4 accumulators = 32 VGPRs).

__global__ void __launch_bounds__(128) fused_kernel(
    const float* __restrict__ x, const float* __restrict__ agg, const float* __restrict__ deg,
    const int* __restrict__ packed,
    const float* __restrict__ bl, const float* __restrict__ b_ih, const float* __restrict__ b_hh,
    const float* __restrict__ b_rec, const float* __restrict__ b_c1, const float* __restrict__ b_c2,
    const float* __restrict__ Wc2,
    float* __restrict__ out_rec, float* __restrict__ out_cls) {
  __shared__ __bf16 sp_lds[4][16][HDIM];
  __shared__ __bf16 h_lds[4][16][HDIM];

  const int wave = threadIdx.x >> 5;
  const int lane = threadIdx.x & 31;
  const int tile = blockIdx.x * 4 + wave;
  if (tile >= (N_NODES / 16)) return;

  const int node_base = tile * 16;
  const int n = lane & 15;   // output column within tile (B/C/D lane index)
  const int hsel = lane >> 4;
  const int node = node_base + n; // row owned by this lane for A-fragments

  const float rdeg = 1.0f / fmaxf(deg[node], 1.0f);
  const float w2a = Wc2[0 * 16 + n], w2b = Wc2[1 * 16 + n];
  const float w2c = Wc2[2 * 16 + n], w2d = Wc2[3 * 16 + n];
  const float b2 = b_c2[0];

  // h0 = 0
  {
    __bf16* hp = &h_lds[wave][0][0];
    for (int i = lane; i < 16 * HDIM; i += 32) hp[i] = (__bf16)0.0f;
  }

#pragma unroll 1
  for (int t = 0; t < T_STEPS; ++t) {
    // ---- A-fragments for x_t and mean-aggregated neighbors (K=32) ----
    v16bf a_x = load_a_g(x + (size_t)node * (T_STEPS * C_IN) + t * C_IN, hsel, 1.0f);
    v16bf a_g = load_a_g(agg + ((size_t)t * N_NODES + node) * C_IN, hsel, rdeg);

    // ---- SAGE: sp = relu(agg@Wl^T + bl + x@Wr^T), staged to LDS as bf16 ----
#pragma unroll 1
    for (int j = 0; j < 8; ++j) {
      float bb = bl[j * 16 + n];
      v8f acc;
#pragma unroll
      for (int r = 0; r < 8; ++r) acc[r] = bb;
      acc = wmma_bf16(a_g, load_frag(packed, FRAG_WL + j, lane), acc);
      acc = wmma_bf16(a_x, load_frag(packed, FRAG_WR + j, lane), acc);
#pragma unroll
      for (int r = 0; r < 8; ++r)
        sp_lds[wave][r + 8 * hsel][j * 16 + n] = (__bf16)fmaxf(acc[r], 0.0f);
    }

    // ---- rebuild A-fragments (K=128) for sp and h ----
    v16bf a_sp[4], a_h[4];
#pragma unroll
    for (int kc = 0; kc < 4; ++kc) {
      a_sp[kc] = load_a_lds(&sp_lds[wave][n][0], hsel, kc);
      a_h[kc]  = load_a_lds(&h_lds[wave][n][0], hsel, kc);
    }

    // ---- GRU gates per 16-column tile of H ----
#pragma unroll 1
    for (int j = 0; j < 8; ++j) {
      float br  = b_ih[0 * HDIM + j * 16 + n] + b_hh[0 * HDIM + j * 16 + n];
      float bz  = b_ih[1 * HDIM + j * 16 + n] + b_hh[1 * HDIM + j * 16 + n];
      float bxn = b_ih[2 * HDIM + j * 16 + n];
      float bhn = b_hh[2 * HDIM + j * 16 + n];
      v8f gr, gz, gxn, ghn;
#pragma unroll
      for (int r = 0; r < 8; ++r) { gr[r] = br; gz[r] = bz; gxn[r] = bxn; ghn[r] = bhn; }
#pragma unroll
      for (int kc = 0; kc < 4; ++kc) {
        gr  = wmma_bf16(a_sp[kc], load_frag(packed, FRAG_WIH + (0 * 8 + j) * 4 + kc, lane), gr);
        gr  = wmma_bf16(a_h[kc],  load_frag(packed, FRAG_WHH + (0 * 8 + j) * 4 + kc, lane), gr);
        gz  = wmma_bf16(a_sp[kc], load_frag(packed, FRAG_WIH + (1 * 8 + j) * 4 + kc, lane), gz);
        gz  = wmma_bf16(a_h[kc],  load_frag(packed, FRAG_WHH + (1 * 8 + j) * 4 + kc, lane), gz);
        gxn = wmma_bf16(a_sp[kc], load_frag(packed, FRAG_WIH + (2 * 8 + j) * 4 + kc, lane), gxn);
        ghn = wmma_bf16(a_h[kc],  load_frag(packed, FRAG_WHH + (2 * 8 + j) * 4 + kc, lane), ghn);
      }
#pragma unroll
      for (int r = 0; r < 8; ++r) {
        float rr = sigmoidf_(gr[r]);
        float zz = sigmoidf_(gz[r]);
        float nv = tanhf(gxn[r] + rr * ghn[r]);
        float hv = (float)h_lds[wave][r + 8 * hsel][j * 16 + n];
        float hn = (1.0f - zz) * nv + zz * hv;
        h_lds[wave][r + 8 * hsel][j * 16 + n] = (__bf16)hn;
      }
    }

    // ---- new h A-fragments for the heads (reuse a_h registers) ----
#pragma unroll
    for (int kc = 0; kc < 4; ++kc) a_h[kc] = load_a_lds(&h_lds[wave][n][0], hsel, kc);

    // ---- reconstruction head: h @ W_rec^T + b_rec -> out[N,T,32] ----
#pragma unroll 1
    for (int j = 0; j < 2; ++j) {
      float bb = b_rec[j * 16 + n];
      v8f acc;
#pragma unroll
      for (int r = 0; r < 8; ++r) acc[r] = bb;
#pragma unroll
      for (int kc = 0; kc < 4; ++kc)
        acc = wmma_bf16(a_h[kc], load_frag(packed, FRAG_WREC + j * 4 + kc, lane), acc);
#pragma unroll
      for (int r = 0; r < 8; ++r) {
        size_t row = (size_t)(node_base + r + 8 * hsel);
        out_rec[(row * T_STEPS + t) * C_OUT + j * 16 + n] = acc[r];
      }
    }

    // ---- classification head: sigmoid(relu(h@Wc1^T+b)@Wc2^T+b2) -> out[N,T] ----
    float part[8];
#pragma unroll
    for (int r = 0; r < 8; ++r) part[r] = 0.0f;
#pragma unroll 1
    for (int j = 0; j < 4; ++j) {
      float bb = b_c1[j * 16 + n];
      float w2 = (j == 0) ? w2a : (j == 1) ? w2b : (j == 2) ? w2c : w2d;
      v8f acc;
#pragma unroll
      for (int r = 0; r < 8; ++r) acc[r] = bb;
#pragma unroll
      for (int kc = 0; kc < 4; ++kc)
        acc = wmma_bf16(a_h[kc], load_frag(packed, FRAG_WC1 + j * 4 + kc, lane), acc);
#pragma unroll
      for (int r = 0; r < 8; ++r) part[r] += fmaxf(acc[r], 0.0f) * w2;
    }
#pragma unroll
    for (int r = 0; r < 8; ++r) {
      float p = part[r];
      p += __shfl_xor(p, 1);
      p += __shfl_xor(p, 2);
      p += __shfl_xor(p, 4);
      p += __shfl_xor(p, 8);
      if (n == 0) {
        size_t row = (size_t)(node_base + r + 8 * hsel);
        out_cls[row * T_STEPS + t] = sigmoidf_(p + b2);
      }
    }
  }
}

extern "C" void kernel_launch(void* const* d_in, const int* in_sizes, int n_in,
                              void* d_out, int out_size, void* d_ws, size_t ws_size,
                              hipStream_t stream) {
  const float*     x    = (const float*)d_in[0];
  const long long* ei   = (const long long*)d_in[1];
  const float*     Wl   = (const float*)d_in[2];
  const float*     bl   = (const float*)d_in[3];
  const float*     Wr   = (const float*)d_in[4];
  const float*     Wih  = (const float*)d_in[5];
  const float*     bih  = (const float*)d_in[6];
  const float*     Whh  = (const float*)d_in[7];
  const float*     bhh  = (const float*)d_in[8];
  const float*     Wrec = (const float*)d_in[9];
  const float*     brec = (const float*)d_in[10];
  const float*     Wc1  = (const float*)d_in[11];
  const float*     bc1  = (const float*)d_in[12];
  const float*     Wc2  = (const float*)d_in[13];
  const float*     bc2  = (const float*)d_in[14];

  // workspace layout: deg[N] | agg[T*N*32] | packed bf16 B-fragments
  float* ws_f   = (float*)d_ws;
  float* deg    = ws_f;
  float* agg    = ws_f + N_NODES;
  int*   packed = (int*)(ws_f + N_NODES + (size_t)T_STEPS * N_NODES * C_IN);

  float* out_rec = (float*)d_out;
  float* out_cls = out_rec + (size_t)N_NODES * T_STEPS * C_OUT;

  long long zcount = (long long)N_NODES + (long long)T_STEPS * N_NODES * C_IN;
  zero_kernel<<<2048, 256, 0, stream>>>(ws_f, zcount);
  deg_kernel<<<(N_EDGES + 255) / 256, 256, 0, stream>>>(ei, deg, N_EDGES);
  agg_kernel<<<(N_EDGES * 4 + 255) / 256, 256, 0, stream>>>(x, ei, agg, N_EDGES);
  prepack_kernel<<<(N_FRAGS * 32 + 255) / 256, 256, 0, stream>>>(Wl, Wr, Wih, Whh, Wrec, Wc1, packed);

  int tiles = N_NODES / 16;                 // 3125
  int grid  = (tiles + 3) / 4;              // 4 waves (tiles) per 128-thread block
  fused_kernel<<<grid, 128, 0, stream>>>(x, agg, deg, packed, bl, bih, bhh, brec,
                                         bc1, bc2, Wc2, out_rec, out_cls);
}
